// SIRModel_30030411333654
// MI455X (gfx1250) — matrix operand order repaired and checked by reading, not compiled, not run on hardware
//
#include <hip/hip_runtime.h>
#include <hip/hip_bf16.h>

typedef __attribute__((ext_vector_type(16))) __bf16 v16bf;
typedef __attribute__((ext_vector_type(8)))  float  v8f;

#define NEG_SLOPE 0.2f

// ---------- helpers ----------
__device__ __forceinline__ unsigned int f2bf(float f) {
  union { float f; unsigned int u; } t; t.f = f;
  unsigned int u = t.u;
  return (u + 0x7FFFu + ((u >> 16) & 1u)) >> 16;   // RNE bf16
}
__device__ __forceinline__ float bf_lo(unsigned int p) {
  union { unsigned int u; float f; } t; t.u = p << 16; return t.f;
}
__device__ __forceinline__ float bf_hi(unsigned int p) {
  union { unsigned int u; float f; } t; t.u = p & 0xFFFF0000u; return t.f;
}
__device__ __forceinline__ v16bf mk16(uint4 a, uint4 b) {
  union { uint4 q[2]; v16bf v; } t; t.q[0] = a; t.q[1] = b; return t.v;
}
__device__ __forceinline__ float lrelu(float x) { return x >= 0.f ? x : NEG_SLOPE * x; }

// ---------- weight packing into WMMA B-fragment order ----------
// pk[((ntile*4+ks)*32+lane)*8+v] = bf16(W[k0,c]) | bf16(W[k0+1,c])<<16
// k0 = ks*32 + (v>=4?16:0) + (lane>=16?8:0) + 2(v&3) ; c = ntile*16 + (lane&15)
__global__ void pack_w(const float* __restrict__ Wq, const float* __restrict__ Wk,
                       const float* __restrict__ Wr, unsigned int* __restrict__ wpk) {
  int t = blockIdx.x * 256 + threadIdx.x;
  if (t >= 12 * 8192) return;
  int wsel = t >> 13;            // 0..11  (= l*3 + type)
  int idx  = t & 8191;
  int l = wsel / 3, type = wsel % 3;
  const float* W = (type == 0 ? Wq : (type == 1 ? Wk : Wr)) + l * 16384;
  int v    = idx & 7;
  int lane = (idx >> 3) & 31;
  int ks   = (idx >> 8) & 3;
  int nt   = idx >> 10;
  int k0 = ks * 32 + ((v & 4) ? 16 : 0) + ((lane & 16) ? 8 : 0) + 2 * (v & 3);
  int c  = nt * 16 + (lane & 15);
  unsigned int lo = f2bf(W[k0 * 128 + c]);
  unsigned int hi = f2bf(W[(k0 + 1) * 128 + c]);
  wpk[t] = lo | (hi << 16);
}

// ---------- atom encoder: h[n,c] = sum_f emb[f, nfeats[n,f], c] ----------
__global__ void atom_enc(const int* __restrict__ nf, const float* __restrict__ emb,
                         float* __restrict__ h, int N) {
  int gid = blockIdx.x * 256 + threadIdx.x;
  if (gid >= N * 128) return;
  int n = gid >> 7, c = gid & 127;
  const int* f = nf + n * 9;
  float acc = 0.f;
#pragma unroll
  for (int j = 0; j < 9; ++j) acc += emb[((j << 6) + f[j]) * 128 + c];
  h[gid] = acc;
}

// ---------- WMMA GEMM (M=32 tile): q = h@Wq + bq ; k = h@Wk + bk  (bf16 out) ----------
__global__ void __launch_bounds__(256) gemm_qk(
    const float* __restrict__ h,
    const unsigned int* __restrict__ wq, const unsigned int* __restrict__ wk,
    const float* __restrict__ bq, const float* __restrict__ bk,
    unsigned short* __restrict__ qo, unsigned short* __restrict__ ko) {
  __shared__ unsigned int AsU[32 * 68];        // 32 rows x 128 bf16, stride 136 bf16 (272B)
  const int row0 = blockIdx.x << 5;
  const int tid = threadIdx.x;
  // stage + f32->bf16 convert the 32x128 activation tile
  for (int i = tid; i < 2048; i += 256) {
    int r = i >> 6, cp = i & 63;
    const float* hp = h + (size_t)(row0 + r) * 128 + cp * 2;
    AsU[r * 68 + cp] = f2bf(hp[0]) | (f2bf(hp[1]) << 16);
  }
  __syncthreads();

  const int lane = tid & 31, w = tid >> 5;
  const int col  = w * 16 + (lane & 15);
  const int r    = lane & 15;
  const int aoff = (lane & 16) ? 4 : 0;        // +8 bf16 for upper half-wave
  v8f aq0 = {}, aq1 = {}, ak0 = {}, ak1 = {};
  const unsigned int* wqp = wq + (size_t)((w * 4) * 32 + lane) * 8;
  const unsigned int* wkp = wk + (size_t)((w * 4) * 32 + lane) * 8;
#pragma unroll
  for (int ks = 0; ks < 4; ++ks) {
    const unsigned int* ap = &AsU[r * 68 + ks * 16 + aoff];
    v16bf A0 = mk16(*(const uint4*)ap, *(const uint4*)(ap + 8));
    const unsigned int* ap1 = ap + 16 * 68;    // rows 16..31 subtile
    v16bf A1 = mk16(*(const uint4*)ap1, *(const uint4*)(ap1 + 8));
    const unsigned int* bp = wqp + ks * 256;
    v16bf Bq = mk16(*(const uint4*)bp, *(const uint4*)(bp + 4));
    bp = wkp + ks * 256;
    v16bf Bk = mk16(*(const uint4*)bp, *(const uint4*)(bp + 4));
    aq0 = __builtin_amdgcn_wmma_f32_16x16x32_bf16(false, A0, false, Bq, (short)0, aq0, false, false);
    aq1 = __builtin_amdgcn_wmma_f32_16x16x32_bf16(false, A1, false, Bq, (short)0, aq1, false, false);
    ak0 = __builtin_amdgcn_wmma_f32_16x16x32_bf16(false, A0, false, Bk, (short)0, ak0, false, false);
    ak1 = __builtin_amdgcn_wmma_f32_16x16x32_bf16(false, A1, false, Bk, (short)0, ak1, false, false);
  }
  const float biasq = bq[col], biask = bk[col];
  const int mbase = (lane & 16) ? 8 : 0;
#pragma unroll
  for (int v = 0; v < 8; ++v) {
    size_t o0 = (size_t)(row0 + mbase + v) * 128 + col;
    size_t o1 = (size_t)(row0 + 16 + mbase + v) * 128 + col;
    qo[o0] = (unsigned short)f2bf(aq0[v] + biasq);
    qo[o1] = (unsigned short)f2bf(aq1[v] + biasq);
    ko[o0] = (unsigned short)f2bf(ak0[v] + biask);
    ko[o1] = (unsigned short)f2bf(ak1[v] + biask);
  }
}

// ---------- WMMA GEMM (M=32 tile): h = lrelu(agg@Wr + br)  (f32 out) ----------
__global__ void __launch_bounds__(256) gemm_r(
    const float* __restrict__ agg, const unsigned int* __restrict__ wr,
    const float* __restrict__ br, float* __restrict__ ho) {
  __shared__ unsigned int AsU[32 * 68];
  const int row0 = blockIdx.x << 5;
  const int tid = threadIdx.x;
  for (int i = tid; i < 2048; i += 256) {
    int r = i >> 6, cp = i & 63;
    const float* hp = agg + (size_t)(row0 + r) * 128 + cp * 2;
    AsU[r * 68 + cp] = f2bf(hp[0]) | (f2bf(hp[1]) << 16);
  }
  __syncthreads();
  const int lane = tid & 31, w = tid >> 5;
  const int col  = w * 16 + (lane & 15);
  const int r    = lane & 15;
  const int aoff = (lane & 16) ? 4 : 0;
  v8f a0 = {}, a1 = {};
  const unsigned int* wrp = wr + (size_t)((w * 4) * 32 + lane) * 8;
#pragma unroll
  for (int ks = 0; ks < 4; ++ks) {
    const unsigned int* ap = &AsU[r * 68 + ks * 16 + aoff];
    v16bf A0 = mk16(*(const uint4*)ap, *(const uint4*)(ap + 8));
    const unsigned int* ap1 = ap + 16 * 68;
    v16bf A1 = mk16(*(const uint4*)ap1, *(const uint4*)(ap1 + 8));
    const unsigned int* bp = wrp + ks * 256;
    v16bf B = mk16(*(const uint4*)bp, *(const uint4*)(bp + 4));
    a0 = __builtin_amdgcn_wmma_f32_16x16x32_bf16(false, A0, false, B, (short)0, a0, false, false);
    a1 = __builtin_amdgcn_wmma_f32_16x16x32_bf16(false, A1, false, B, (short)0, a1, false, false);
  }
  const float bias = br[col];
  const int mbase = (lane & 16) ? 8 : 0;
#pragma unroll
  for (int v = 0; v < 8; ++v) {
    ho[(size_t)(row0 + mbase + v) * 128 + col]      = lrelu(a0[v] + bias);
    ho[(size_t)(row0 + 16 + mbase + v) * 128 + col] = lrelu(a1[v] + bias);
  }
}

// ---------- edge message + segment-sum via f32 atomics ----------
__global__ void edge_k(const unsigned short* __restrict__ q, const unsigned short* __restrict__ k,
                       const int* __restrict__ src, const int* __restrict__ dst,
                       float* __restrict__ agg, int E) {
  int gi = blockIdx.x * 256 + threadIdx.x;   // E*16 = 16M fits in int
  int e = gi >> 4;
  if (e >= E) return;
  int part = gi & 15;
  int s = src[e], d = dst[e];
  const uint4 qv = *(const uint4*)(q + (size_t)d * 128 + part * 8);
  const uint4 kv = *(const uint4*)(k + (size_t)s * 128 + part * 8);
  float* out = agg + (size_t)d * 128 + part * 8;
  const unsigned int qa[4] = {qv.x, qv.y, qv.z, qv.w};
  const unsigned int ka[4] = {kv.x, kv.y, kv.z, kv.w};
#pragma unroll
  for (int i = 0; i < 4; ++i) {
    float m0 = lrelu(bf_lo(qa[i]) + bf_lo(ka[i]));
    float m1 = lrelu(bf_hi(qa[i]) + bf_hi(ka[i]));
    unsafeAtomicAdd(out + 2 * i, m0);
    unsafeAtomicAdd(out + 2 * i + 1, m1);
  }
}

// ---------- graph sum-pooling ----------
__global__ void pool_k(const float* __restrict__ h, const int* __restrict__ gids,
                       float* __restrict__ hg, int N) {
  int gi = blockIdx.x * 256 + threadIdx.x;
  int n = gi >> 4;
  if (n >= N) return;
  int part = gi & 15;
  int g = gids[n];
  const float4 v0 = *(const float4*)(h + (size_t)n * 128 + part * 8);
  const float4 v1 = *(const float4*)(h + (size_t)n * 128 + part * 8 + 4);
  float* out = hg + (size_t)g * 128 + part * 8;
  unsafeAtomicAdd(out + 0, v0.x); unsafeAtomicAdd(out + 1, v0.y);
  unsafeAtomicAdd(out + 2, v0.z); unsafeAtomicAdd(out + 3, v0.w);
  unsafeAtomicAdd(out + 4, v1.x); unsafeAtomicAdd(out + 5, v1.y);
  unsafeAtomicAdd(out + 6, v1.z); unsafeAtomicAdd(out + 7, v1.w);
}

// ---------- utility ----------
__global__ void zero_f(float* p, int n4) {
  int g = blockIdx.x * 256 + threadIdx.x;
  if (g < n4) ((float4*)p)[g] = make_float4(0.f, 0.f, 0.f, 0.f);
}

// ---------- tiny MLP head ----------
__global__ void mlp(const float* __restrict__ x, const float* __restrict__ W,
                    const float* __restrict__ b, float* __restrict__ y,
                    int G, int K, int C) {
  int gid = blockIdx.x * 256 + threadIdx.x;
  if (gid >= G * C) return;
  int g = gid / C, c = gid % C;
  float acc = b[c];
  for (int t = 0; t < K; ++t) acc += x[(size_t)g * K + t] * W[t * C + c];
  y[gid] = acc;
}

__global__ void bnstats(const float* __restrict__ x, int G, int C,
                        float* __restrict__ mu, float* __restrict__ rstd) {
  __shared__ float s1[256], s2[256];
  int c = blockIdx.x;
  float a = 0.f, b = 0.f;
  for (int g = threadIdx.x; g < G; g += 256) {
    float v = x[(size_t)g * C + c];
    a += v; b += v * v;
  }
  s1[threadIdx.x] = a; s2[threadIdx.x] = b;
  __syncthreads();
  for (int s = 128; s > 0; s >>= 1) {
    if (threadIdx.x < s) { s1[threadIdx.x] += s1[threadIdx.x + s]; s2[threadIdx.x] += s2[threadIdx.x + s]; }
    __syncthreads();
  }
  if (threadIdx.x == 0) {
    float m = s1[0] / (float)G;
    float var = s2[0] / (float)G - m * m;
    mu[c] = m;
    rstd[c] = rsqrtf(var + 1e-5f);
  }
}

__global__ void bnapply(float* __restrict__ x, const float* __restrict__ mu,
                        const float* __restrict__ rstd, const float* __restrict__ gm,
                        const float* __restrict__ bt, int n, int C) {
  int gid = blockIdx.x * 256 + threadIdx.x;
  if (gid >= n) return;
  int c = gid % C;
  x[gid] = lrelu((x[gid] - mu[c]) * rstd[c] * gm[c] + bt[c]);
}

// ---------- host launch ----------
extern "C" void kernel_launch(void* const* d_in, const int* in_sizes, int n_in,
                              void* d_out, int out_size, void* d_ws, size_t ws_size,
                              hipStream_t stream) {
  const int N = in_sizes[0] / 9;     // 100000
  const int E = in_sizes[2];         // 1000000
  const int G = out_size;            // 1000
  const int H = 128, L = 4;

  const int*   nfeats = (const int*)d_in[0];
  const int*   src    = (const int*)d_in[2];
  const int*   dst    = (const int*)d_in[3];
  const int*   gids   = (const int*)d_in[4];
  const float* emb    = (const float*)d_in[5];
  const float* Wq = (const float*)d_in[6];
  const float* bq = (const float*)d_in[7];
  const float* Wk = (const float*)d_in[8];
  const float* bk = (const float*)d_in[9];
  const float* Wr = (const float*)d_in[10];
  const float* br = (const float*)d_in[11];
  const float* W1 = (const float*)d_in[12];
  const float* b1 = (const float*)d_in[13];
  const float* g1 = (const float*)d_in[14];
  const float* be1 = (const float*)d_in[15];
  const float* W2 = (const float*)d_in[16];
  const float* b2 = (const float*)d_in[17];
  const float* g2 = (const float*)d_in[18];
  const float* be2 = (const float*)d_in[19];
  const float* W3 = (const float*)d_in[20];
  const float* b3 = (const float*)d_in[21];
  float* out = (float*)d_out;

  // workspace carve-up
  size_t off = 0;
  char* base = (char*)d_ws;
  auto carve = [&](size_t bytes) { void* p = base + off; off += (bytes + 255) & ~(size_t)255; return p; };
  float*          h   = (float*)carve((size_t)N * H * 4);
  float*          agg = (float*)carve((size_t)N * H * 4);
  unsigned short* q   = (unsigned short*)carve((size_t)N * H * 2);
  unsigned short* k   = (unsigned short*)carve((size_t)N * H * 2);
  unsigned int*   wpk = (unsigned int*)carve((size_t)12 * 8192 * 4);
  float*          hg  = (float*)carve((size_t)G * H * 4);
  float*          x1  = (float*)carve((size_t)G * 64 * 4);
  float*          x2  = (float*)carve((size_t)G * 32 * 4);
  float*          mu1 = (float*)carve(64 * 4);
  float*          rs1 = (float*)carve(64 * 4);
  float*          mu2 = (float*)carve(32 * 4);
  float*          rs2 = (float*)carve(32 * 4);
  (void)ws_size; (void)n_in;

  // 1) pack weights into WMMA fragment order (bf16)
  pack_w<<<(12 * 8192 + 255) / 256, 256, 0, stream>>>(Wq, Wk, Wr, wpk);

  // 2) atom encoder
  atom_enc<<<(N * 128 + 255) / 256, 256, 0, stream>>>(nfeats, emb, h, N);

  // 3) message-passing layers
  const int tiles = N / 32;                 // N divisible by 32
  for (int l = 0; l < L; ++l) {
    const unsigned int* wq_l = wpk + (size_t)(l * 3 + 0) * 8192;
    const unsigned int* wk_l = wpk + (size_t)(l * 3 + 1) * 8192;
    const unsigned int* wr_l = wpk + (size_t)(l * 3 + 2) * 8192;
    gemm_qk<<<tiles, 256, 0, stream>>>(h, wq_l, wk_l, bq + l * H, bk + l * H, q, k);
    zero_f<<<(N * H / 4 + 255) / 256, 256, 0, stream>>>(agg, N * H / 4);
    edge_k<<<(E * 16 + 255) / 256, 256, 0, stream>>>(q, k, src, dst, agg, E);
    gemm_r<<<tiles, 256, 0, stream>>>(agg, wr_l, br + l * H, h);
  }

  // 4) graph pooling
  zero_f<<<(G * H / 4 + 255) / 256, 256, 0, stream>>>(hg, G * H / 4);
  pool_k<<<(N * 16 + 255) / 256, 256, 0, stream>>>(h, gids, hg, N);

  // 5) MLP head: fc1 -> bn -> lrelu -> fc2 -> bn -> lrelu -> fc3
  mlp<<<(G * 64 + 255) / 256, 256, 0, stream>>>(hg, W1, b1, x1, G, 128, 64);
  bnstats<<<64, 256, 0, stream>>>(x1, G, 64, mu1, rs1);
  bnapply<<<(G * 64 + 255) / 256, 256, 0, stream>>>(x1, mu1, rs1, g1, be1, G * 64, 64);
  mlp<<<(G * 32 + 255) / 256, 256, 0, stream>>>(x1, W2, b2, x2, G, 64, 32);
  bnstats<<<32, 256, 0, stream>>>(x2, G, 32, mu2, rs2);
  bnapply<<<(G * 32 + 255) / 256, 256, 0, stream>>>(x2, mu2, rs2, g2, be2, G * 32, 32);
  mlp<<<(G + 255) / 256, 256, 0, stream>>>(x2, W3, b3, out, G, 32, 1);
}
